// T5Attention_45234595561596
// MI455X (gfx1250) — matrix operand order repaired and checked by reading, not compiled, not run on hardware
//
#include <hip/hip_runtime.h>
#include <math.h>

// ---------- types ----------
typedef __bf16 bf16;
typedef __attribute__((ext_vector_type(16))) __bf16 v16bf;
typedef __attribute__((ext_vector_type(8)))  __bf16 v8bf;
typedef __attribute__((ext_vector_type(4)))  __bf16 v4bf;
typedef __attribute__((ext_vector_type(8)))  float  v8f;

#define WMMA_BF16(a, b, c) \
  __builtin_amdgcn_wmma_f32_16x16x32_bf16(false, (a), false, (b), (short)0, (c), false, false)

static __device__ __forceinline__ v8f zero_v8f() {
  v8f z;
#pragma unroll
  for (int i = 0; i < 8; ++i) z[i] = 0.0f;
  return z;
}

// ---------- WMMA fragment loaders (gfx1250 wave32 layouts, CDNA5 ISA) ----------
// A fragment 16x32 bf16: lane<16 row m holds K 0..7 (v0-3) & 16..23 (v4-7);
// lane>=16 same row holds K 8..15 & 24..31.  p = &A[row0][k0], row stride lda.
static __device__ __forceinline__ v16bf frag_a(const bf16* __restrict__ p, int lda, int lane) {
  const int m    = lane & 15;
  const int koff = (lane >> 4) << 3;  // 0 or 8
  const v8bf lo = *(const v8bf*)(p + (size_t)m * lda + koff);
  const v8bf hi = *(const v8bf*)(p + (size_t)m * lda + 16 + koff);
  v16bf f;
#pragma unroll
  for (int i = 0; i < 8; ++i) { f[i] = lo[i]; f[8 + i] = hi[i]; }
  return f;
}

// B fragment 32x16 bf16: lane<16 column n holds K 0..7 & 16..23; lane>=16 holds K 8..15 & 24..31.
// Sourced from a buffer where each B-column is contiguous in K (i.e. B^T row-major).
// p = &BT[n0][k0], row stride ldk.  Two 16-byte loads per lane.
static __device__ __forceinline__ v16bf frag_b(const bf16* __restrict__ p, int ldk, int lane) {
  const int n  = lane & 15;
  const int kb = (lane >> 4) << 3;  // 0 or 8
  const v8bf lo = *(const v8bf*)(p + (size_t)n * ldk + kb);
  const v8bf hi = *(const v8bf*)(p + (size_t)n * ldk + 16 + kb);
  v16bf f;
#pragma unroll
  for (int i = 0; i < 8; ++i) { f[i] = lo[i]; f[8 + i] = hi[i]; }
  return f;
}

// A fragment from an fp32 LDS row-block (softmax probs) -> bf16, via float4 LDS loads
// so the backend can emit ds_load_b128 + packed cvt.
static __device__ __forceinline__ v16bf frag_a_lds(const float* __restrict__ P, int ldr,
                                                   int k0, int lane) {
  const int m    = lane & 15;
  const int koff = (lane >> 4) << 3;
  const float4* lo = (const float4*)(P + (size_t)m * ldr + k0 + koff);       // 16B aligned
  const float4* hi = (const float4*)(P + (size_t)m * ldr + k0 + 16 + koff);
  const float4 l0 = lo[0], l1 = lo[1], h0 = hi[0], h1 = hi[1];
  v16bf f;
  f[0]  = (bf16)l0.x; f[1]  = (bf16)l0.y; f[2]  = (bf16)l0.z; f[3]  = (bf16)l0.w;
  f[4]  = (bf16)l1.x; f[5]  = (bf16)l1.y; f[6]  = (bf16)l1.z; f[7]  = (bf16)l1.w;
  f[8]  = (bf16)h0.x; f[9]  = (bf16)h0.y; f[10] = (bf16)h0.z; f[11] = (bf16)h0.w;
  f[12] = (bf16)h1.x; f[13] = (bf16)h1.y; f[14] = (bf16)h1.z; f[15] = (bf16)h1.w;
  return f;
}

// ---------- T5 relative-position bucket (bidirectional; nb=16, max_exact=8, maxdist=128) ----------
static __device__ __forceinline__ int rel_bucket(int rel) {
  const int rb = (rel > 0) ? 16 : 0;
  int ar = (rel < 0) ? -rel : rel;
  const int arc = (ar < 1) ? 1 : ar;
  int lg = 8 + (int)(__logf((float)arc * 0.125f) * (8.0f / 2.7725887f));  // log(16)
  lg = (lg < 15) ? lg : 15;
  return rb + ((ar < 8) ? ar : lg);
}

// ---------- elementwise fp32 -> bf16 (x4 vectorized) ----------
__global__ void f32_to_bf16_k(const float* __restrict__ src, bf16* __restrict__ dst, int n4) {
  int i = blockIdx.x * blockDim.x + threadIdx.x;
  if (i < n4) {
    const float4 v = ((const float4*)src)[i];
    v4bf o = { (bf16)v.x, (bf16)v.y, (bf16)v.z, (bf16)v.w };
    ((v4bf*)dst)[i] = o;
  }
}

// ---------- tiled transpose + convert: W[K][N] fp32 -> WT[N][K] bf16 ----------
__global__ __launch_bounds__(256) void transpose_f32_to_bf16_k(const float* __restrict__ W,
                                                               bf16* __restrict__ WT,
                                                               int K, int N) {
  __shared__ float t[32][33];
  const int tiles_n = N >> 5;
  const int bx = blockIdx.x % tiles_n;
  const int by = blockIdx.x / tiles_n;
  const int tx = threadIdx.x & 31;
  const int ty = threadIdx.x >> 5;
#pragma unroll
  for (int i = ty; i < 32; i += 8)
    t[i][tx] = W[(size_t)(by * 32 + i) * N + bx * 32 + tx];
  __syncthreads();
#pragma unroll
  for (int i = ty; i < 32; i += 8)
    WT[(size_t)(bx * 32 + i) * K + by * 32 + tx] = (bf16)t[tx][i];
}

// ---------- bf16 WMMA GEMM: C[4096x1024] = A * B, B given as BT[N][K] ----------
// One wave computes a 32x64 block: 2 A-fragments x 4 B-fragments -> 8 WMMAs per 32-K step.
// MODE 0: store bf16 in [b,h,s,d]      (Q and K for attention)
// MODE 2: store bf16 in [b,h,d,s]      (V transposed -> contiguous B-fragments in flash)
// MODE 1: store fp32 row-major [row][col] (final output projection)
template <int MODE>
__global__ __launch_bounds__(256) void gemm_bf16_wmma(const bf16* __restrict__ A,
                                                      const bf16* __restrict__ BT,
                                                      void* __restrict__ out) {
  constexpr int N = 1024, K = 1024;
  constexpr int NSTRIPS = N / 64;  // 16
  const int lane = threadIdx.x & 31;
  const int wg   = blockIdx.x * 8 + (threadIdx.x >> 5);
  const int m0   = (wg / NSTRIPS) * 32;
  const int n0   = (wg % NSTRIPS) * 64;

  v8f acc[2][4];
#pragma unroll
  for (int r = 0; r < 2; ++r)
#pragma unroll
    for (int t = 0; t < 4; ++t) acc[r][t] = zero_v8f();

  for (int kk = 0; kk < K; kk += 32) {
    const v16bf a0 = frag_a(A + (size_t)m0 * K + kk, K, lane);
    const v16bf a1 = frag_a(A + (size_t)(m0 + 16) * K + kk, K, lane);
#pragma unroll
    for (int t = 0; t < 4; ++t) {
      const v16bf b = frag_b(BT + (size_t)(n0 + t * 16) * K + kk, K, lane);
      acc[0][t] = WMMA_BF16(a0, b, acc[0][t]);
      acc[1][t] = WMMA_BF16(a1, b, acc[1][t]);
    }
  }

  const int n  = lane & 15;
  const int mb = (lane >> 4) << 3;
#pragma unroll
  for (int r = 0; r < 2; ++r) {
#pragma unroll
    for (int t = 0; t < 4; ++t) {
      const int col = n0 + t * 16 + n;
#pragma unroll
      for (int j = 0; j < 8; ++j) {
        const int row = m0 + r * 16 + mb + j;
        const float v = acc[r][t][j];
        if constexpr (MODE == 1) {
          ((float*)out)[(size_t)row * N + col] = v;
        } else {
          const int b_ = row >> 11, s = row & 2047;  // S=2048
          const int h  = col >> 6,  d = col & 63;    // DK=64
          if constexpr (MODE == 0)
            ((bf16*)out)[(((size_t)b_ * 16 + h) * 2048 + s) * 64 + d] = (bf16)v;
          else  // MODE == 2: V transposed [b,h,d,s]
            ((bf16*)out)[(((size_t)b_ * 16 + h) * 64 + d) * 2048 + s] = (bf16)v;
        }
      }
    }
  }
}

// ---------- fused flash attention: one block = one (b,h, 16-row q-tile) ----------
// 4 waves, 128 threads.  Score strip 16x2048 fp32 + 4096-entry bias LUT live in
// LDS (~149 KB -> 2 workgroups per 320 KB WGP).  Softmax passes vectorized x4.
__global__ __launch_bounds__(128) void t5_flash_attn(const bf16* __restrict__ Q,
                                                     const bf16* __restrict__ Kc,
                                                     const bf16* __restrict__ VT,
                                                     const float* __restrict__ bias_tab,
                                                     bf16* __restrict__ attn) {
  constexpr int S = 2048, DK = 64, H = 16, SLD = 2064;  // +16 pad vs LDS banks
  extern __shared__ float smem[];
  float* P      = smem;                // 16 * SLD
  float* red    = smem + 16 * SLD;     // 128
  float* rowmax = red + 128;           // 16
  float* rowsum = rowmax + 16;         // 16
  float* btab   = rowsum + 16;         // 32 : bias_tab[bucket][h] for this head
  float* lut    = btab + 32;           // 4096: bias value per relative position

  const int bh   = blockIdx.x >> 7;    // b*16 + h
  const int qt   = blockIdx.x & 127;
  const int h    = bh & 15;
  const int q0   = qt * 16;
  const int wave = threadIdx.x >> 5;
  const int lane = threadIdx.x & 31;
  const int n    = lane & 15;
  const int mb   = (lane >> 4) << 3;

  // stage per-head bias column, then expand into a relative-position LUT:
  // lut[idx] = bias(rel = idx - 2048); needed idx range is [1, 4095].
  if (threadIdx.x < 32) btab[threadIdx.x] = bias_tab[threadIdx.x * H + h];
  __syncthreads();
  for (int i = threadIdx.x; i < 4096; i += 128)
    lut[i] = btab[rel_bucket(i - 2048)];

  // ---- Phase 1: S_tile = Q * K^T via WMMA, stored to LDS ----
  const bf16* Qb = Q + ((size_t)bh * S + q0) * DK;
  const v16bf qa0 = frag_a(Qb, DK, lane);        // d = 0..31
  const v16bf qa1 = frag_a(Qb + 32, DK, lane);   // d = 32..63
  const bf16* Kb = Kc + (size_t)bh * S * DK;
  for (int kt = wave; kt < S / 16; kt += 4) {
    const bf16* Kt = Kb + (size_t)kt * 16 * DK;  // rows of K == columns of B
    v16bf b0 = frag_b(Kt, DK, lane);
    v16bf b1 = frag_b(Kt + 32, DK, lane);
    v8f c = zero_v8f();
    c = WMMA_BF16(qa0, b0, c);
    c = WMMA_BF16(qa1, b1, c);
#pragma unroll
    for (int j = 0; j < 8; ++j)
      P[(size_t)(mb + j) * SLD + kt * 16 + n] = c[j];
  }
  __syncthreads();

  // ---- Phase 2: bias + softmax per row (8 threads per row, float4 vectorized) ----
  const int r   = threadIdx.x >> 3;
  const int sub = threadIdx.x & 7;
  const int qg  = q0 + r;
  float* Pr = P + (size_t)r * SLD;
  const float* Lr = lut + 2048 - qg;   // Lr[c] == bias(c - qg)

  float mx = -3.0e38f;
  for (int c0 = sub * 4; c0 < S; c0 += 32) {
    float4 v = *(float4*)(Pr + c0);
    v.x += Lr[c0 + 0];
    v.y += Lr[c0 + 1];
    v.z += Lr[c0 + 2];
    v.w += Lr[c0 + 3];
    *(float4*)(Pr + c0) = v;
    mx = fmaxf(mx, fmaxf(fmaxf(v.x, v.y), fmaxf(v.z, v.w)));
  }
  red[threadIdx.x] = mx;
  __syncthreads();
  if (sub == 0) {
    float m2 = red[r * 8];
#pragma unroll
    for (int i = 1; i < 8; ++i) m2 = fmaxf(m2, red[r * 8 + i]);
    rowmax[r] = m2;
  }
  __syncthreads();
  const float rm = rowmax[r];
  float sum = 0.0f;
  for (int c0 = sub * 4; c0 < S; c0 += 32) {
    float4 v = *(float4*)(Pr + c0);
    v.x = __expf(v.x - rm);
    v.y = __expf(v.y - rm);
    v.z = __expf(v.z - rm);
    v.w = __expf(v.w - rm);
    *(float4*)(Pr + c0) = v;
    sum += (v.x + v.y) + (v.z + v.w);
  }
  red[threadIdx.x] = sum;
  __syncthreads();
  if (sub == 0) {
    float s2 = 0.0f;
#pragma unroll
    for (int i = 0; i < 8; ++i) s2 += red[r * 8 + i];
    rowsum[r] = s2;
  }
  __syncthreads();

  // ---- Phase 3: O = P * V via WMMA; each wave owns one 16-wide d-slice ----
  const int n0 = wave * 16;
  const bf16* Vb = VT + (size_t)bh * DK * S;   // [DK][S] slice -> contiguous B frags
  v8f acc = zero_v8f();
  for (int kk = 0; kk < S; kk += 32) {
    const v16bf a = frag_a_lds(P, SLD, kk, lane);
    const v16bf b = frag_b(Vb + (size_t)n0 * S + kk, S, lane);
    acc = WMMA_BF16(a, b, acc);
  }
  // store bf16 attn_out at [b, s, h*DK + d]; fold 1/rowsum normalization here
  bf16* dst = attn + (((size_t)(bh >> 4)) * S + q0) * (H * DK) + h * DK + n0;
#pragma unroll
  for (int j = 0; j < 8; ++j) {
    const float v = acc[j] / rowsum[mb + j];
    dst[(size_t)(mb + j) * (H * DK) + n] = (bf16)v;
  }
}

// ---------- host launcher ----------
extern "C" void kernel_launch(void* const* d_in, const int* in_sizes, int n_in,
                              void* d_out, int out_size, void* d_ws, size_t ws_size,
                              hipStream_t stream) {
  (void)in_sizes; (void)n_in; (void)out_size; (void)ws_size;

  const float* X        = (const float*)d_in[0];  // [2,2048,1024]
  const float* Wq       = (const float*)d_in[1];  // [1024,1024]
  const float* Wk       = (const float*)d_in[2];
  const float* Wv       = (const float*)d_in[3];
  const float* Wo       = (const float*)d_in[4];
  const float* bias_tab = (const float*)d_in[5];  // [32,16]

  constexpr size_t M = 4096, D = 1024;                 // rows = B*S, model dim
  constexpr size_t QKV_ELEMS = 2ull * 16 * 2048 * 64;  // B*H*S*DK

  char* ws = (char*)d_ws;
  size_t off = 0;
  bf16* Xb   = (bf16*)(ws + off); off += M * D * 2;         // 8 MB
  bf16* WqT  = (bf16*)(ws + off); off += D * D * 2;         // 2 MB
  bf16* WkT  = (bf16*)(ws + off); off += D * D * 2;
  bf16* WvT  = (bf16*)(ws + off); off += D * D * 2;
  bf16* WoT  = (bf16*)(ws + off); off += D * D * 2;
  bf16* Qb   = (bf16*)(ws + off); off += QKV_ELEMS * 2;     // 8 MB each
  bf16* Kb   = (bf16*)(ws + off); off += QKV_ELEMS * 2;
  bf16* Vt   = (bf16*)(ws + off); off += QKV_ELEMS * 2;     // [b,h,d,s]
  bf16* attn = (bf16*)(ws + off); off += M * D * 2;         // 8 MB

  // 1) convert activations, transpose-convert weights to bf16
  {
    const int n4 = (int)(M * D / 4);
    f32_to_bf16_k<<<(n4 + 255) / 256, 256, 0, stream>>>(X, Xb, n4);
    transpose_f32_to_bf16_k<<<1024, 256, 0, stream>>>(Wq, WqT, 1024, 1024);
    transpose_f32_to_bf16_k<<<1024, 256, 0, stream>>>(Wk, WkT, 1024, 1024);
    transpose_f32_to_bf16_k<<<1024, 256, 0, stream>>>(Wv, WvT, 1024, 1024);
    transpose_f32_to_bf16_k<<<1024, 256, 0, stream>>>(Wo, WoT, 1024, 1024);
  }

  // 2) Q/K/V projections (4096x1024x1024 bf16 WMMA GEMMs, 32x64 per wave)
  gemm_bf16_wmma<0><<<256, 256, 0, stream>>>(Xb, WqT, (void*)Qb);
  gemm_bf16_wmma<0><<<256, 256, 0, stream>>>(Xb, WkT, (void*)Kb);
  gemm_bf16_wmma<2><<<256, 256, 0, stream>>>(Xb, WvT, (void*)Vt);  // transposed V

  // 3) fused flash attention: 2*16*128 blocks, ~149.2 KB dynamic LDS each
  {
    const size_t shbytes = (16 * 2064 + 128 + 16 + 16 + 32 + 4096) * sizeof(float);
    (void)hipFuncSetAttribute(reinterpret_cast<const void*>(t5_flash_attn),
                              hipFuncAttributeMaxDynamicSharedMemorySize, (int)shbytes);
    t5_flash_attn<<<4096, 128, shbytes, stream>>>(Qb, Kb, Vt, bias_tab, attn);
  }

  // 4) output projection -> fp32 d_out
  gemm_bf16_wmma<1><<<256, 256, 0, stream>>>(attn, WoT, d_out);
}
